// VariableSelectionNetwork_36696200577751
// MI455X (gfx1250) — matrix operand order repaired and checked by reading, not compiled, not run on hardware
//
#include <hip/hip_runtime.h>

// ---------------------------------------------------------------------------
// VariableSelectionNetwork fused forward for gfx1250 (MI455X, wave32, WMMA)
// B=8 S=512 F=32 H=256 C=128 ; N=B*S=4096 ; FH=8192
// All contractions run on v_wmma_f32_16x16x32_bf16 (bf16 in, f32 accumulate).
// Weights are pre-transposed to [nout][K] bf16 so WMMA B fragments are two
// ds_load_b128's per tile (no u16 gathers). LDS tiles padded to 40 halves
// (20 dwords) per row -> conflict-free fragment loads.
// ---------------------------------------------------------------------------

#define NTOK 4096   // B*S
#define HID  256
#define NF   32
#define FH   8192
#define CTXD 128
#define SEQ  512
#define PADK 40     // padded K-stride (halves) for LDS tiles

typedef __bf16 bf16_t;
typedef bf16_t v16bf __attribute__((ext_vector_type(16)));
typedef float  v8f   __attribute__((ext_vector_type(8)));

union ABFrag { unsigned short u[16]; uint4 q[2]; v16bf v; };

__device__ __forceinline__ unsigned short f2bf(float f) {
  unsigned u = __float_as_uint(f);
  u += 0x7fffu + ((u >> 16) & 1u);           // round-to-nearest-even
  return (unsigned short)(u >> 16);
}
__device__ __forceinline__ float eluf(float x) { return x > 0.f ? x : (expf(x) - 1.f); }
__device__ __forceinline__ float sigm(float x) { return 1.f / (1.f + expf(-x)); }

// ---------------------------------------------------------------------------
// f32 [Z][K][nout] -> bf16 transposed [Z][nout][K]  (32x32 LDS tile transpose)
// ---------------------------------------------------------------------------
__global__ __launch_bounds__(256) void cvt_transpose_bf16(
    const float* __restrict__ src, unsigned short* __restrict__ dst, int K, int nout) {
  __shared__ unsigned short t[32][33];
  const int k0 = blockIdx.x * 32, j0 = blockIdx.y * 32, z = blockIdx.z;
  const float* s = src + (size_t)z * K * nout;
  unsigned short* d = dst + (size_t)z * K * nout;

  const int kr = threadIdx.x >> 3;          // 0..31
  const int jc = (threadIdx.x & 7) * 4;     // 0..28
  float4 v = *(const float4*)(s + (size_t)(k0 + kr) * nout + j0 + jc);
  t[kr][jc + 0] = f2bf(v.x);
  t[kr][jc + 1] = f2bf(v.y);
  t[kr][jc + 2] = f2bf(v.z);
  t[kr][jc + 3] = f2bf(v.w);
  __syncthreads();

  const int jr = threadIdx.x >> 3;          // 0..31
  const int kc = (threadIdx.x & 7) * 4;     // 0..28
  unsigned short o[4];
  o[0] = t[kc + 0][jr];
  o[1] = t[kc + 1][jr];
  o[2] = t[kc + 2][jr];
  o[3] = t[kc + 3][jr];
  *(uint2*)(d + (size_t)(j0 + jr) * K + k0 + kc) = *(const uint2*)o;
}

// ---------------------------------------------------------------------------
// Generic WMMA GEMM: out = act(A[N,K] @ W[K,nout] + bias), W given as Wt[nout][K].
// Block tile 64x64, 8 waves (4 row-waves x 2 col-waves of 16x32), K chunk 32.
// amode: 0 = A is f32 buffer (lda)        -> bf16 on the fly
//        1 = A[n][k] = elu(x[n,z]*w1[z,k]+b1[z,k])   (stage-1 GRN, z=feature)
//        2 = A[n][k] = hs[n,k] + ctxh[n/SEQ, k]      (selection GRN layer 2)
// ---------------------------------------------------------------------------
__global__ __launch_bounds__(256) void gemm_bf16_wmma(
    const float* __restrict__ Af32, int lda,
    const unsigned short* __restrict__ Wt, long wZ,
    const float* __restrict__ bias, long biasZ,
    int K, int nout,
    unsigned short* __restrict__ outB, float* __restrict__ outF, int ldo, long outZ,
    int amode, int act,
    const float* __restrict__ xp, const float* __restrict__ w1p, const float* __restrict__ b1p,
    const float* __restrict__ ctxh) {
  __shared__ __align__(16) unsigned short As[64 * PADK];   // [64 rows][32 K] padded
  __shared__ __align__(16) unsigned short Ws[64 * PADK];   // [64 cols][32 K] padded

  const int tid = threadIdx.x;
  const int lane = tid & 31;
  const int wid = tid >> 5;
  const int wave_m = wid & 3;   // 0..3 -> 16-row slice
  const int wave_n = wid >> 2;  // 0..1 -> 32-col slice
  const int n0 = blockIdx.x * 64;
  const int j0 = blockIdx.y * 64;
  const int z = blockIdx.z;

  const unsigned short* Wz = Wt + (size_t)z * wZ;
  const float* biasz = bias + (size_t)z * biasZ;

  v8f c0 = {0.f, 0.f, 0.f, 0.f, 0.f, 0.f, 0.f, 0.f};
  v8f c1 = c0;

  const int ar = tid >> 2;        // 0..63 (A row)
  const int ak = (tid & 3) * 8;   // A k group
  const int wr = tid >> 2;        // 0..63 (W col row in Wt)
  const int wk = (tid & 3) * 8;   // W k group

  for (int kc = 0; kc < K; kc += 32) {
    { // stage A tile (64x32 bf16)
      const int n = n0 + ar;
      float v[8];
      if (amode == 0) {
        const float4* pa = (const float4*)(Af32 + (size_t)n * lda + kc + ak);
        float4 a0 = pa[0], a1 = pa[1];
        v[0] = a0.x; v[1] = a0.y; v[2] = a0.z; v[3] = a0.w;
        v[4] = a1.x; v[5] = a1.y; v[6] = a1.z; v[7] = a1.w;
      } else if (amode == 1) {
        const float xv = xp[(size_t)n * NF + z];
#pragma unroll
        for (int i = 0; i < 8; i++) {
          const int h = kc + ak + i;
          v[i] = eluf(xv * w1p[(size_t)z * HID + h] + b1p[(size_t)z * HID + h]);
        }
      } else {
        const int bidx = n >> 9;  // n / SEQ
#pragma unroll
        for (int i = 0; i < 8; i++) {
          const int h = kc + ak + i;
          v[i] = Af32[(size_t)n * HID + h] + ctxh[bidx * HID + h];
        }
      }
      unsigned short o[8];
#pragma unroll
      for (int i = 0; i < 8; i++) o[i] = f2bf(v[i]);
      *(uint4*)(As + ar * PADK + ak) = *(const uint4*)o;
    }
    { // stage W tile from Wt[nout][K]: 64 cols x 32 K, zero-pad cols >= nout
      const int j = j0 + wr;
      uint4 wv = {0u, 0u, 0u, 0u};
      if (j < nout) wv = *(const uint4*)(Wz + (size_t)j * K + kc + wk);
      *(uint4*)(Ws + wr * PADK + wk) = wv;
    }
    __syncthreads();

    // A fragment: M = lane%16, K = (lane/16)*16 + i  (contiguous 16 halves)
    ABFrag af;
    {
      const int arow = wave_m * 16 + (lane & 15);
      const int kb = (lane >> 4) * 16;
      af.q[0] = *(const uint4*)(As + arow * PADK + kb);
      af.q[1] = *(const uint4*)(As + arow * PADK + kb + 8);
    }
    // B fragments: N = lane%16, K = (lane/16)*16 + i  (contiguous 16 halves)
    const int kb2 = (lane >> 4) * 16;
    const int col0 = wave_n * 32 + (lane & 15);
    ABFrag bf0, bf1;
    bf0.q[0] = *(const uint4*)(Ws + col0 * PADK + kb2);
    bf0.q[1] = *(const uint4*)(Ws + col0 * PADK + kb2 + 8);
    bf1.q[0] = *(const uint4*)(Ws + (col0 + 16) * PADK + kb2);
    bf1.q[1] = *(const uint4*)(Ws + (col0 + 16) * PADK + kb2 + 8);

    c0 = __builtin_amdgcn_wmma_f32_16x16x32_bf16(false, af.v, false, bf0.v, (short)0, c0, false, false);
    c1 = __builtin_amdgcn_wmma_f32_16x16x32_bf16(false, af.v, false, bf1.v, (short)0, c1, false, false);
    __syncthreads();
  }

  // epilogue: D element (m,n): n = lane%16, m = 8*(lane/16)+r
  const int mrow = wave_m * 16 + (lane >> 4) * 8;
  const int jc0 = j0 + wave_n * 32 + (lane & 15);
#pragma unroll
  for (int r = 0; r < 8; r++) {
    const int n = n0 + mrow + r;
    int j = jc0;
    if (j < nout) {
      float vv = c0[r] + biasz[j];
      if (act) vv = eluf(vv);
      if (outF) outF[(size_t)n * ldo + j] = vv;
      if (outB) outB[(size_t)z * outZ + (size_t)n * ldo + j] = f2bf(vv);
    }
    j = jc0 + 16;
    if (j < nout) {
      float vv = c1[r] + biasz[j];
      if (act) vv = eluf(vv);
      if (outF) outF[(size_t)n * ldo + j] = vv;
      if (outB) outB[(size_t)z * outZ + (size_t)n * ldo + j] = f2bf(vv);
    }
  }
}

// ---------------------------------------------------------------------------
// Fused gates GEMM + GLU + skip + LayerNorm.  W given as Wt[512][256].
// g = A[32,256] @ W[256,512] + bias, then per row:
//   t = g[:256]*sigmoid(g[256:]) + skip ; out = LN(t)*gamma + beta
// skipmode 0: skip = x[n,z]*ws[z,:]+bs[z,:]; out -> stacked[n][z][:]  (f32)
// skipmode 1: skip = combined[n,:];          out -> final d_out[n][:]
// Block: 32 rows x 512 cols, 8 waves (2 row x 4 col waves of 16x128).
// ---------------------------------------------------------------------------
__global__ __launch_bounds__(256) void gluln_wmma(
    const void* __restrict__ Abase, long aZ, int abf16,
    const unsigned short* __restrict__ Wt, long wZ,
    const float* __restrict__ bias, long biasZ,
    int skipmode,
    const float* __restrict__ xp, const float* __restrict__ wsp, const float* __restrict__ bsp,
    const float* __restrict__ comb,
    const float* __restrict__ gamma, long gammaZ,
    const float* __restrict__ beta, long betaZ,
    float* __restrict__ outStacked, float* __restrict__ outFinal) {
  __shared__ __align__(16) unsigned char smem[32 * 512 * 4];  // 64 KB, aliased
  unsigned short* As = (unsigned short*)smem;                 // [32][PADK]   (loop,  2.5 KB)
  unsigned short* Ws = (unsigned short*)(smem + 32 * PADK * 2); // [512][PADK] (loop, 40 KB)
  float* gates = (float*)smem;                                // [32][512]    (epilogue)

  const int tid = threadIdx.x;
  const int lane = tid & 31;
  const int wid = tid >> 5;
  const int wave_m = wid & 1;
  const int wave_n = wid >> 1;  // 0..3, 128 cols each
  const int n0 = blockIdx.x * 32;
  const int z = blockIdx.y;

  const unsigned short* Wz = Wt + (size_t)z * wZ;
  const float* biasz = bias + (size_t)z * biasZ;

  v8f c[8];
#pragma unroll
  for (int i = 0; i < 8; i++) c[i] = (v8f){0.f, 0.f, 0.f, 0.f, 0.f, 0.f, 0.f, 0.f};

  const int ar = tid >> 3;        // 0..31
  const int ak = (tid & 7) * 4;   // 0..28

  for (int kc = 0; kc < HID; kc += 32) {
    { // A tile 32x32
      const int n = n0 + ar;
      if (abf16) {
        const unsigned short* pa =
            (const unsigned short*)Abase + (size_t)z * aZ + (size_t)n * HID + kc + ak;
        *(uint2*)(As + ar * PADK + ak) = *(const uint2*)pa;
      } else {
        const float* pa = (const float*)Abase + (size_t)n * HID + kc + ak;
        float4 a0 = *(const float4*)pa;
        unsigned short o[4];
        o[0] = f2bf(a0.x); o[1] = f2bf(a0.y); o[2] = f2bf(a0.z); o[3] = f2bf(a0.w);
        *(uint2*)(As + ar * PADK + ak) = *(const uint2*)o;
      }
    }
    // W tile from Wt[512][256]: 512 cols x 32 K, coalesced uint4
#pragma unroll
    for (int i = 0; i < 8; i++) {
      const int idx = tid + i * 256;      // 0..2047
      const int j = idx >> 2;             // 0..511
      const int kk = (idx & 3) * 8;       // 0,8,16,24
      *(uint4*)(Ws + j * PADK + kk) = *(const uint4*)(Wz + (size_t)j * HID + kc + kk);
    }
    // prefetch next W chunk into cache (global_prefetch_b8)
    if (kc + 32 < HID)
      __builtin_prefetch((const char*)Wz + (size_t)tid * 512 + (kc + 32) * 2, 0, 1);
    __syncthreads();

    ABFrag af;
    {
      const int arow = wave_m * 16 + (lane & 15);
      const int kb = (lane >> 4) * 16;
      af.q[0] = *(const uint4*)(As + arow * PADK + kb);
      af.q[1] = *(const uint4*)(As + arow * PADK + kb + 8);
    }
    const int kb2 = (lane >> 4) * 16;
#pragma unroll
    for (int tt = 0; tt < 8; tt++) {
      ABFrag bfx;
      const int col = wave_n * 128 + tt * 16 + (lane & 15);
      bfx.q[0] = *(const uint4*)(Ws + col * PADK + kb2);
      bfx.q[1] = *(const uint4*)(Ws + col * PADK + kb2 + 8);
      c[tt] = __builtin_amdgcn_wmma_f32_16x16x32_bf16(false, af.v, false, bfx.v, (short)0,
                                                      c[tt], false, false);
    }
    __syncthreads();
  }

  // spill gates (+bias) to LDS (aliases loop buffers; last barrier passed)
  {
    const int mbase = wave_m * 16 + (lane >> 4) * 8;
#pragma unroll
    for (int tt = 0; tt < 8; tt++) {
      const int j = wave_n * 128 + tt * 16 + (lane & 15);
      const float bv = biasz[j];
#pragma unroll
      for (int r = 0; r < 8; r++) gates[(mbase + r) * 512 + j] = c[tt][r] + bv;
    }
  }
  __syncthreads();

  // GLU + skip + LayerNorm: 4 rows per wave, one h-strip (8 vals) per lane
#pragma unroll
  for (int rr = 0; rr < 4; rr++) {
    const int m = wid * 4 + rr;
    const int n = n0 + m;
    float t[8];
    float s = 0.f, sq = 0.f;
    float xv = 0.f;
    if (skipmode == 0) xv = xp[(size_t)n * NF + z];
#pragma unroll
    for (int i = 0; i < 8; i++) {
      const int h = i * 32 + lane;
      const float gv = gates[m * 512 + h];
      const float gs = gates[m * 512 + 256 + h];
      const float glu = gv * sigm(gs);
      const float sk = (skipmode == 0)
                           ? (xv * wsp[(size_t)z * HID + h] + bsp[(size_t)z * HID + h])
                           : comb[(size_t)n * HID + h];
      t[i] = glu + sk;
      s += t[i];
      sq += t[i] * t[i];
    }
#pragma unroll
    for (int off = 16; off > 0; off >>= 1) {
      s += __shfl_xor(s, off, 32);
      sq += __shfl_xor(sq, off, 32);
    }
    const float mean = s * (1.f / 256.f);
    const float var = sq * (1.f / 256.f) - mean * mean;
    const float rstd = rsqrtf(var + 1e-5f);
#pragma unroll
    for (int i = 0; i < 8; i++) {
      const int h = i * 32 + lane;
      const float y = (t[i] - mean) * rstd * gamma[(size_t)z * gammaZ + h] +
                      beta[(size_t)z * betaZ + h];
      if (skipmode == 0)
        outStacked[((size_t)n * NF + z) * HID + h] = y;
      else
        outFinal[(size_t)n * HID + h] = y;
    }
  }
}

// ---------------------------------------------------------------------------
// ctxh[b][h] = context[b] @ sctx[:,h]   (8 x 256, K=128 -> trivial)
// ---------------------------------------------------------------------------
__global__ __launch_bounds__(256) void vsn_ctx(const float* __restrict__ ctx,
                                               const float* __restrict__ sctx,
                                               float* __restrict__ ctxh) {
  const int b = blockIdx.x, h = threadIdx.x;
  float acc = 0.f;
  for (int cc = 0; cc < CTXD; cc++) acc += ctx[b * CTXD + cc] * sctx[cc * HID + h];
  ctxh[b * HID + h] = acc;
}

// ---------------------------------------------------------------------------
// weights = softmax(LN(glu2 + sskip)) over F=32: one wave per row, lane = f
// ---------------------------------------------------------------------------
__global__ __launch_bounds__(256) void vsn_weights(
    const float* __restrict__ gg, const float* __restrict__ sskip,
    const float* __restrict__ sg, const float* __restrict__ sbt,
    float* __restrict__ wbuf, float* __restrict__ out_w) {
  const int lane = threadIdx.x & 31;
  const int wid = threadIdx.x >> 5;
  const int n = blockIdx.x * 8 + wid;
  const float a = gg[(size_t)n * 64 + lane];
  const float b = gg[(size_t)n * 64 + 32 + lane];
  const float t = a * sigm(b) + sskip[(size_t)n * NF + lane];
  float s = t, sq = t * t;
#pragma unroll
  for (int off = 16; off > 0; off >>= 1) {
    s += __shfl_xor(s, off, 32);
    sq += __shfl_xor(sq, off, 32);
  }
  const float mean = s * (1.f / 32.f);
  const float var = sq * (1.f / 32.f) - mean * mean;
  const float ln = (t - mean) * rsqrtf(var + 1e-5f) * sg[lane] + sbt[lane];
  float mx = ln;
#pragma unroll
  for (int off = 16; off > 0; off >>= 1) mx = fmaxf(mx, __shfl_xor(mx, off, 32));
  const float e = expf(ln - mx);
  float se = e;
#pragma unroll
  for (int off = 16; off > 0; off >>= 1) se += __shfl_xor(se, off, 32);
  const float w = e / se;
  wbuf[(size_t)n * NF + lane] = w;
  out_w[(size_t)n * NF + lane] = w;
}

// ---------------------------------------------------------------------------
// combined[n][h] = sum_f stacked[n][f][h] * w[n][f]
// ---------------------------------------------------------------------------
__global__ __launch_bounds__(256) void vsn_combine(const float* __restrict__ stacked,
                                                   const float* __restrict__ wbuf,
                                                   float* __restrict__ comb) {
  __shared__ float wl[NF];
  const int n = blockIdx.x, h = threadIdx.x;
  if (h < NF) wl[h] = wbuf[(size_t)n * NF + h];
  __syncthreads();
  float acc = 0.f;
#pragma unroll
  for (int f = 0; f < NF; f++) acc += stacked[((size_t)n * NF + f) * HID + h] * wl[f];
  comb[(size_t)n * HID + h] = acc;
}

// ---------------------------------------------------------------------------
extern "C" void kernel_launch(void* const* d_in, const int* in_sizes, int n_in,
                              void* d_out, int out_size, void* d_ws, size_t ws_size,
                              hipStream_t stream) {
  const float* x    = (const float*)d_in[0];
  const float* ctx  = (const float*)d_in[1];
  const float* w1   = (const float*)d_in[2];
  const float* b1   = (const float*)d_in[3];
  const float* w2   = (const float*)d_in[4];
  const float* b2   = (const float*)d_in[5];
  const float* wg   = (const float*)d_in[6];
  const float* bg   = (const float*)d_in[7];
  const float* wsk  = (const float*)d_in[8];
  const float* bs   = (const float*)d_in[9];
  const float* g1   = (const float*)d_in[10];
  const float* be1  = (const float*)d_in[11];
  const float* sw1  = (const float*)d_in[12];
  const float* sb1  = (const float*)d_in[13];
  const float* sctx = (const float*)d_in[14];
  const float* sw2  = (const float*)d_in[15];
  const float* sb2  = (const float*)d_in[16];
  const float* sgw  = (const float*)d_in[17];
  const float* sgb  = (const float*)d_in[18];
  const float* sg   = (const float*)d_in[19];
  const float* sbt  = (const float*)d_in[20];
  const float* skw  = (const float*)d_in[21];
  const float* skb  = (const float*)d_in[22];
  const float* ow1  = (const float*)d_in[23];
  const float* ob1  = (const float*)d_in[24];
  const float* ow2  = (const float*)d_in[25];
  const float* ob2  = (const float*)d_in[26];
  const float* ogw  = (const float*)d_in[27];
  const float* ogb  = (const float*)d_in[28];
  const float* og   = (const float*)d_in[29];
  const float* obt  = (const float*)d_in[30];

  char* base = (char*)d_ws;
  size_t off = 0;
  auto carve = [&](size_t bytes) -> char* {
    char* r = base + off;
    off = (off + bytes + 255) & ~((size_t)255);
    return r;
  };
  unsigned short* w2b  = (unsigned short*)carve((size_t)2 * NF * HID * HID);
  unsigned short* wgb  = (unsigned short*)carve((size_t)2 * NF * HID * 512);
  unsigned short* sw1b = (unsigned short*)carve((size_t)2 * FH * HID);
  unsigned short* skwb = (unsigned short*)carve((size_t)2 * FH * NF);
  unsigned short* sw2b = (unsigned short*)carve((size_t)2 * HID * HID);
  unsigned short* sgwb = (unsigned short*)carve((size_t)2 * HID * 64);
  unsigned short* ow1b = (unsigned short*)carve((size_t)2 * HID * HID);
  unsigned short* ow2b = (unsigned short*)carve((size_t)2 * HID * HID);
  unsigned short* ogwb = (unsigned short*)carve((size_t)2 * HID * 512);
  unsigned short* h2   = (unsigned short*)carve((size_t)2 * NF * NTOK * HID);
  float* stacked = (float*)carve((size_t)4 * NTOK * NF * HID);
  float* hs      = (float*)carve((size_t)4 * NTOK * HID);
  float* ctxh    = (float*)carve((size_t)4 * 8 * HID);
  float* hs2     = (float*)carve((size_t)4 * NTOK * HID);
  float* gg      = (float*)carve((size_t)4 * NTOK * 64);
  float* sskip   = (float*)carve((size_t)4 * NTOK * NF);
  float* wts     = (float*)carve((size_t)4 * NTOK * NF);
  float* comb    = (float*)carve((size_t)4 * NTOK * HID);
  float* ho1     = (float*)carve((size_t)4 * NTOK * HID);
  float* ho2     = (float*)carve((size_t)4 * NTOK * HID);

  float* out_final = (float*)d_out;                      // [N,256]
  float* out_w     = out_final + (size_t)NTOK * HID;     // [N,32]

  // Convert + transpose all weight matrices to bf16 Wt[nout][K]
  auto cvtT = [&](const float* s, unsigned short* d, int K, int nout, int Z) {
    cvt_transpose_bf16<<<dim3(K / 32, nout / 32, Z), 256, 0, stream>>>(s, d, K, nout);
  };
  cvtT(w2,  w2b,  HID, HID, NF);
  cvtT(wg,  wgb,  HID, 512, NF);
  cvtT(sw1, sw1b, FH,  HID, 1);
  cvtT(skw, skwb, FH,  NF,  1);
  cvtT(sw2, sw2b, HID, HID, 1);
  cvtT(sgw, sgwb, HID, 64,  1);
  cvtT(ow1, ow1b, HID, HID, 1);
  cvtT(ow2, ow2b, HID, HID, 1);
  cvtT(ogw, ogwb, HID, 512, 1);

  // Stage 1: h2[f] = elu( elu(x*w1+b1) @ w2[f] + b2[f] )   (A generated on the fly)
  gemm_bf16_wmma<<<dim3(NTOK / 64, HID / 64, NF), 256, 0, stream>>>(
      nullptr, 0, w2b, (long)HID * HID, b2, (long)HID, HID, HID,
      h2, nullptr, HID, (long)NTOK * HID, /*amode=*/1, /*act=*/1, x, w1, b1, nullptr);

  // context projection
  vsn_ctx<<<8, HID, 0, stream>>>(ctx, sctx, ctxh);

  // Stage 2: gates = h2[f]@wg[f]+bg[f]; GLU+skip+LN -> stacked[n][f][h]
  gluln_wmma<<<dim3(NTOK / 32, NF), 256, 0, stream>>>(
      h2, (long)NTOK * HID, /*abf16=*/1, wgb, (long)HID * 512, bg, (long)512,
      /*skipmode=*/0, x, wsk, bs, nullptr, g1, (long)HID, be1, (long)HID,
      stacked, nullptr);

  // Selection GRN: hs = elu(flat@sw1+sb1)
  gemm_bf16_wmma<<<dim3(NTOK / 64, HID / 64, 1), 256, 0, stream>>>(
      stacked, FH, sw1b, 0, sb1, 0, FH, HID,
      nullptr, hs, HID, 0, /*amode=*/0, /*act=*/1, nullptr, nullptr, nullptr, nullptr);

  // hs2 = elu((hs+ctxh)@sw2+sb2)
  gemm_bf16_wmma<<<dim3(NTOK / 64, HID / 64, 1), 256, 0, stream>>>(
      hs, HID, sw2b, 0, sb2, 0, HID, HID,
      nullptr, hs2, HID, 0, /*amode=*/2, /*act=*/1, nullptr, nullptr, nullptr, ctxh);

  // gg = hs2@sgw+sgb  [N,64]
  gemm_bf16_wmma<<<dim3(NTOK / 64, 1, 1), 256, 0, stream>>>(
      hs2, HID, sgwb, 0, sgb, 0, HID, 64,
      nullptr, gg, 64, 0, /*amode=*/0, /*act=*/0, nullptr, nullptr, nullptr, nullptr);

  // sskip = flat@skw+skb  [N,32]
  gemm_bf16_wmma<<<dim3(NTOK / 64, 1, 1), 256, 0, stream>>>(
      stacked, FH, skwb, 0, skb, 0, FH, NF,
      nullptr, sskip, NF, 0, /*amode=*/0, /*act=*/0, nullptr, nullptr, nullptr, nullptr);

  // weights = softmax(LN(glu2+sskip))  (also written straight into d_out)
  vsn_weights<<<NTOK / 8, 256, 0, stream>>>(gg, sskip, sg, sbt, wts, out_w);

  // combined = sum_f stacked * weights
  vsn_combine<<<NTOK, HID, 0, stream>>>(stacked, wts, comb);

  // Output GRN: ho1 = elu(comb@ow1+ob1); ho2 = elu(ho1@ow2+ob2)
  gemm_bf16_wmma<<<dim3(NTOK / 64, HID / 64, 1), 256, 0, stream>>>(
      comb, HID, ow1b, 0, ob1, 0, HID, HID,
      nullptr, ho1, HID, 0, /*amode=*/0, /*act=*/1, nullptr, nullptr, nullptr, nullptr);
  gemm_bf16_wmma<<<dim3(NTOK / 64, HID / 64, 1), 256, 0, stream>>>(
      ho1, HID, ow2b, 0, ob2, 0, HID, HID,
      nullptr, ho2, HID, 0, /*amode=*/0, /*act=*/1, nullptr, nullptr, nullptr, nullptr);

  // go = ho2@ogw+ogb; GLU + combined skip + LN -> final output
  gluln_wmma<<<dim3(NTOK / 32, 1), 256, 0, stream>>>(
      ho2, 0, /*abf16=*/0, ogwb, 0, ogb, 0,
      /*skipmode=*/1, nullptr, nullptr, nullptr, comb, og, 0, obt, 0,
      nullptr, out_final);
}
